// CapsuleLayer_6124623364155
// MI455X (gfx1250) — compile-verified, matrix operand order checked by reading, not run on hardware
//
#include <hip/hip_runtime.h>
#include <hip/hip_bf16.h>
#include <stdint.h>

// ---------------------------------------------------------------------------
// CapsuleLayer routing for MI455X (gfx1250, wave32, WMMA).
//
// x(B=64,N=2048,Din=16) f32, W(C=32,N=2048,D=32,Din=16) f32 -> out(B,C,D) f32.
//
// W -> bf16 WMMA-B fragments (64 MiB, L2-resident), x -> bf16 WMMA-A
// fragments (4 MiB). inputs_hat is never materialized: recomputed in
// registers via v_wmma_f32_16x16x32_bf16. s-GEMM contracts over (n,i) with
// two n per K=32 (no K padding); routing weights applied with v_pk_mul_bf16
// (inline CDNA5 asm). b-update reduces over d with a level-interleaved
// shfl_xor butterfly (8 independent chains in flight per DS wait).
// ---------------------------------------------------------------------------

typedef __attribute__((ext_vector_type(16))) __bf16        v16bf;
typedef __attribute__((ext_vector_type(8)))  float         v8f;
typedef __attribute__((ext_vector_type(8)))  unsigned int  v8u;

static constexpr int Bb = 64;    // batch
static constexpr int Nn = 2048;  // input capsules
static constexpr int Di = 16;    // input dim
static constexpr int Cc = 32;    // output capsules
static constexpr int Dd = 32;    // output dim
static constexpr int NCH = 32;   // n per workgroup in GEMM passes

// RNE float pair -> packed bf16 dword
__device__ __forceinline__ unsigned pkbf(float a, float b) {
  unsigned ua = __float_as_uint(a), ub = __float_as_uint(b);
  unsigned ha = (ua + 0x7FFFu + ((ua >> 16) & 1u)) >> 16;
  unsigned hb = (ub + 0x7FFFu + ((ub >> 16) & 1u)) >> 16;
  return (ha & 0xFFFFu) | (hb << 16);
}

// packed 2xbf16 multiply: V_PK_MUL_BF16 (VOP3P op 42) -- no clang builtin
__device__ __forceinline__ unsigned pk_mul_bf16(unsigned a, unsigned b) {
  unsigned d;
  asm("v_pk_mul_bf16 %0, %1, %2" : "=v"(d) : "v"(a), "v"(b));
  return d;
}

__device__ __forceinline__ v8f wmma_bf16(const v8u& A, const v8u& B, v8f acc) {
  v16bf a = __builtin_bit_cast(v16bf, A);
  v16bf b = __builtin_bit_cast(v16bf, B);
  // 8 args: (neg_a, A, neg_b, B, c_mod, C, reuse_a, reuse_b)
  return __builtin_amdgcn_wmma_f32_16x16x32_bf16(false, a, false, b,
                                                 (short)0, acc, false, false);
}

// ---------------------------------------------------------------------------
// Pack x(B,N,Di) f32 -> bf16 A-fragment order (16B/lane).
// Per (n,bt): lane l<16 -> b=bt*16+l, i=0..7; lane l>=16 -> b, i=8..15.
// xpk[(n*4+bt)*32 + lane] : uint4 (8 bf16)
// ---------------------------------------------------------------------------
__global__ void __launch_bounds__(256)
k_pack_x(const float* __restrict__ x, uint4* __restrict__ xpk) {
  int t = blockIdx.x * blockDim.x + threadIdx.x;  // < N*4*32
  int lane = t & 31, bt = (t >> 5) & 3, n = t >> 7;
  int b  = bt * 16 + (lane & 15);
  int i0 = (lane >> 4) * 8;
  const float4* src = (const float4*)(x + ((size_t)b * Nn + n) * Di + i0);
  float4 a = src[0], c = src[1];
  uint4 o;
  o.x = pkbf(a.x, a.y); o.y = pkbf(a.z, a.w);
  o.z = pkbf(c.x, c.y); o.w = pkbf(c.z, c.w);
  xpk[t] = o;
}

// ---------------------------------------------------------------------------
// Pack W(C,N,D,Di) f32 -> bf16 B-fragment order (16B/lane).
// Per (c,n,dt): lane l<16 -> d=dt*16+l, i=0..7; lane l>=16 -> d, i=8..15.
// wpk[((c*N+n)*2+dt)*32 + lane] : uint4
// ---------------------------------------------------------------------------
__global__ void __launch_bounds__(256)
k_pack_w(const float* __restrict__ W, uint4* __restrict__ wpk) {
  int t = blockIdx.x * blockDim.x + threadIdx.x;  // < C*N*2*32
  int lane = t & 31, dt = (t >> 5) & 1, cn = t >> 6;  // cn = c*N + n
  int d  = dt * 16 + (lane & 15);
  int i0 = (lane >> 4) * 8;
  const float4* src = (const float4*)(W + ((size_t)cn * Dd + d) * Di + i0);
  float4 a = src[0], b = src[1];
  uint4 o;
  o.x = pkbf(a.x, a.y); o.y = pkbf(a.z, a.w);
  o.z = pkbf(b.x, b.y); o.w = pkbf(b.z, b.w);
  wpk[t] = o;
}

__global__ void __launch_bounds__(256)
k_zero(float* __restrict__ p, int nElem) {
  int t = blockIdx.x * blockDim.x + threadIdx.x;
  if (t < nElem) p[t] = 0.0f;
}

// ---------------------------------------------------------------------------
// s[b,c,d] += sum_n cw[b,c,n] * hat[b,c,n,d]
//           = sum_{(n,i)} (cw*x)[b,n,i] * W[c,n,d,i]   (K = (n,i), 2 n / WMMA)
// Grid: C * (N/NCH) blocks, 8 waves; wave -> (bt = w&3, dt = w>>2).
// UNIFORM=true: first iteration, weight 1/C folded into the epilogue.
// ---------------------------------------------------------------------------
template <bool UNIFORM>
__global__ void __launch_bounds__(256)
k_sgemm(const uint4* __restrict__ xpk, const uint4* __restrict__ wpk,
        const float* __restrict__ cw, float* __restrict__ s) {
  int wave = threadIdx.x >> 5, lane = threadIdx.x & 31;
  int bt = wave & 3, dt = wave >> 2;
  int nblocks = Nn / NCH;
  int c = blockIdx.x / nblocks;
  int nbase = (blockIdx.x % nblocks) * NCH;
  int bl = bt * 16 + (lane & 15);

  v8f acc = {};
#pragma unroll 2
  for (int np = 0; np < NCH; np += 2) {
    int n0 = nbase + np, n1 = n0 + 1;
    // speculative stream-prefetch of the next fragments (global_prefetch_b8)
    __builtin_prefetch(&wpk[((size_t)c * Nn + n0 + 2) * 64 + dt * 32 + lane], 0, 1);
    __builtin_prefetch(&xpk[(size_t)((n0 + 2) * 4 + bt) * 32 + lane], 0, 1);

    uint4 xv0 = xpk[(size_t)(n0 * 4 + bt) * 32 + lane];
    uint4 xv1 = xpk[(size_t)(n1 * 4 + bt) * 32 + lane];
    v8u A;
    if constexpr (UNIFORM) {
      A[0] = xv0.x; A[1] = xv0.y; A[2] = xv0.z; A[3] = xv0.w;
      A[4] = xv1.x; A[5] = xv1.y; A[6] = xv1.z; A[7] = xv1.w;
    } else {
      float w0 = cw[((size_t)c * Nn + n0) * 64 + bl];
      float w1 = cw[((size_t)c * Nn + n1) * 64 + bl];
      unsigned wp0 = pkbf(w0, w0), wp1 = pkbf(w1, w1);
      A[0] = pk_mul_bf16(xv0.x, wp0); A[1] = pk_mul_bf16(xv0.y, wp0);
      A[2] = pk_mul_bf16(xv0.z, wp0); A[3] = pk_mul_bf16(xv0.w, wp0);
      A[4] = pk_mul_bf16(xv1.x, wp1); A[5] = pk_mul_bf16(xv1.y, wp1);
      A[6] = pk_mul_bf16(xv1.z, wp1); A[7] = pk_mul_bf16(xv1.w, wp1);
    }
    uint4 wv0 = wpk[((size_t)c * Nn + n0) * 64 + dt * 32 + lane];
    uint4 wv1 = wpk[((size_t)c * Nn + n1) * 64 + dt * 32 + lane];
    v8u Bf;
    Bf[0] = wv0.x; Bf[1] = wv0.y; Bf[2] = wv0.z; Bf[3] = wv0.w;
    Bf[4] = wv1.x; Bf[5] = wv1.y; Bf[6] = wv1.z; Bf[7] = wv1.w;
    acc = wmma_bf16(A, Bf, acc);
  }

  const float scale = UNIFORM ? (1.0f / 32.0f) : 1.0f;
  int dcol = dt * 16 + (lane & 15);
  int half = lane >> 4;
#pragma unroll
  for (int r = 0; r < 8; ++r) {
    int brow = bt * 16 + r + 8 * half;  // WMMA C-layout row
    atomicAdd(&s[((size_t)brow * Cc + c) * Dd + dcol], acc[r] * scale);
  }
}

// ---------------------------------------------------------------------------
// bbuf[b,c,n] += sum_d out[b,c,d] * hat[b,c,n,d]
// hat tile per n via one WMMA (K padded 16->32 with zeros).
// d-reduction: level-interleaved shfl_xor butterfly (8 chains in flight per
// level -> 4 DS waits per n), then one predicated clause of 8 atomics.
// ---------------------------------------------------------------------------
__global__ void __launch_bounds__(256)
k_bupd(const uint4* __restrict__ xpk, const uint4* __restrict__ wpk,
       const float* __restrict__ out, float* __restrict__ bbuf) {
  int wave = threadIdx.x >> 5, lane = threadIdx.x & 31;
  int bt = wave & 3, dt = wave >> 2;
  int nblocks = Nn / NCH;
  int c = blockIdx.x / nblocks;
  int nbase = (blockIdx.x % nblocks) * NCH;
  int dcol = dt * 16 + (lane & 15);
  int half = lane >> 4;

  float ov[8];  // out[b,c,d] operand is n-invariant: hoist out of the loop
#pragma unroll
  for (int r = 0; r < 8; ++r) {
    int brow = bt * 16 + r + 8 * half;
    ov[r] = out[((size_t)brow * Cc + c) * Dd + dcol];
  }

  for (int n = nbase; n < nbase + NCH; ++n) {
    __builtin_prefetch(&wpk[((size_t)c * Nn + n + 1) * 64 + dt * 32 + lane], 0, 1);
    uint4 xv = xpk[(size_t)(n * 4 + bt) * 32 + lane];
    v8u A;
    A[0] = xv.x; A[1] = xv.y; A[2] = xv.z; A[3] = xv.w;
    A[4] = 0u;   A[5] = 0u;   A[6] = 0u;   A[7] = 0u;
    uint4 wv = wpk[((size_t)c * Nn + n) * 64 + dt * 32 + lane];
    v8u Bf;
    Bf[0] = wv.x; Bf[1] = wv.y; Bf[2] = wv.z; Bf[3] = wv.w;
    Bf[4] = 0u;  Bf[5] = 0u;  Bf[6] = 0u;  Bf[7] = 0u;
    v8f acc = {};
    acc = wmma_bf16(A, Bf, acc);  // hat tile (16b x 16d) for this (c,n)

    // p[r] = sum over the 16 d-lanes of each half of acc[r]*ov[r]
    float p[8];
#pragma unroll
    for (int r = 0; r < 8; ++r) p[r] = acc[r] * ov[r];
#pragma unroll
    for (int m = 1; m <= 8; m <<= 1) {
      float t[8];
#pragma unroll
      for (int r = 0; r < 8; ++r) t[r] = __shfl_xor(p[r], m, 32);
#pragma unroll
      for (int r = 0; r < 8; ++r) p[r] += t[r];
    }
    if ((lane & 15) == 0) {  // lanes 0 and 16 carry the two half-sums
      size_t base = ((size_t)c * Nn + n) * 64 + bt * 16 + 8 * half;
#pragma unroll
      for (int r = 0; r < 8; ++r) atomicAdd(&bbuf[base + r], p[r]);
    }
  }
}

// softmax over C for each (n,b); layouts bbuf/cw = [c][n][b]
__global__ void __launch_bounds__(256)
k_softmax(const float* __restrict__ bbuf, float* __restrict__ cw) {
  int t = blockIdx.x * blockDim.x + threadIdx.x;  // < N*64
  const size_t NB = (size_t)Nn * 64;
  float v[Cc], m = -1e30f;
#pragma unroll
  for (int c = 0; c < Cc; ++c) { v[c] = bbuf[(size_t)c * NB + t]; m = fmaxf(m, v[c]); }
  float sum = 0.0f;
#pragma unroll
  for (int c = 0; c < Cc; ++c) { v[c] = __expf(v[c] - m); sum += v[c]; }
  float inv = 1.0f / sum;
#pragma unroll
  for (int c = 0; c < Cc; ++c) cw[(size_t)c * NB + t] = v[c] * inv;
}

// squash over D (one wave per (b,c) group); optionally re-zero s for next pass
__global__ void __launch_bounds__(256)
k_squash(float* __restrict__ s, float* __restrict__ out, int zero_s) {
  int t = blockIdx.x * blockDim.x + threadIdx.x;  // < B*C*D, lane == d
  float v = s[t];
  float sq = v * v;
  sq += __shfl_xor(sq, 1, 32);
  sq += __shfl_xor(sq, 2, 32);
  sq += __shfl_xor(sq, 4, 32);
  sq += __shfl_xor(sq, 8, 32);
  sq += __shfl_xor(sq, 16, 32);
  float scale = sq / (1.0f + sq) * rsqrtf(sq + 1e-7f);
  out[t] = scale * v;
  if (zero_s) s[t] = 0.0f;
}

extern "C" void kernel_launch(void* const* d_in, const int* in_sizes, int n_in,
                              void* d_out, int out_size, void* d_ws, size_t ws_size,
                              hipStream_t stream) {
  const float* x = (const float*)d_in[0];   // (64,2048,16)
  const float* W = (const float*)d_in[1];   // (32,2048,32,16)
  float* outp = (float*)d_out;              // (64,32,32)

  // Workspace carve-up (~100.5 MiB total)
  char*  ws   = (char*)d_ws;
  uint4* xpk  = (uint4*)ws;                                    //  4 MiB
  uint4* wpk  = (uint4*)(ws + (size_t)(4u << 20));             // 64 MiB
  float* bbuf = (float*)(ws + (size_t)(68u << 20));            // 16 MiB
  float* cw   = bbuf + (size_t)Cc * Nn * 64;                   // 16 MiB
  float* sbuf = cw + (size_t)Cc * Nn * 64;                     // 256 KiB
  float* obuf = sbuf + (size_t)Bb * Cc * Dd;                   // 256 KiB

  // One-time (per call) packing + accumulator init
  k_pack_x<<<Nn * 4 * 32 / 256, 256, 0, stream>>>(x, xpk);
  k_pack_w<<<Cc * Nn * 2 * 32 / 256, 256, 0, stream>>>(W, wpk);
  k_zero<<<(Cc * Nn * 64) / 256, 256, 0, stream>>>(bbuf, Cc * Nn * 64);
  k_zero<<<(Bb * Cc * Dd) / 256, 256, 0, stream>>>(sbuf, Bb * Cc * Dd);

  const int gemm_grid = Cc * (Nn / NCH);  // 2048 workgroups

  // Routing iteration 0: c uniform = 1/C (folded into epilogue scale)
  k_sgemm<true><<<gemm_grid, 256, 0, stream>>>(xpk, wpk, nullptr, sbuf);
  k_squash<<<(Bb * Cc * Dd) / 256, 256, 0, stream>>>(sbuf, obuf, 1);
  k_bupd<<<gemm_grid, 256, 0, stream>>>(xpk, wpk, obuf, bbuf);
  k_softmax<<<(Nn * 64) / 256, 256, 0, stream>>>(bbuf, cw);

  // Routing iteration 1
  k_sgemm<false><<<gemm_grid, 256, 0, stream>>>(xpk, wpk, cw, sbuf);
  k_squash<<<(Bb * Cc * Dd) / 256, 256, 0, stream>>>(sbuf, obuf, 1);
  k_bupd<<<gemm_grid, 256, 0, stream>>>(xpk, wpk, obuf, bbuf);
  k_softmax<<<(Nn * 64) / 256, 256, 0, stream>>>(bbuf, cw);

  // Routing iteration 2 (final): squash straight into d_out
  k_sgemm<false><<<gemm_grid, 256, 0, stream>>>(xpk, wpk, cw, sbuf);
  k_squash<<<(Bb * Cc * Dd) / 256, 256, 0, stream>>>(sbuf, outp, 0);
}
